// MultiHeadAttention_48782238548272
// MI455X (gfx1250) — matrix operand order repaired and checked by reading, not compiled, not run on hardware
//
#include <hip/hip_runtime.h>
#include <hip/hip_bf16.h>

#define HIDDEN   1024
#define HEADS    16
#define HEAD_DIM 64
#define BATCH    4
#define SEQ      2048
#define MROWS    (BATCH * SEQ)          // 8192
#define EROW     2056                   // padded LDS row stride (floats), 32B-aligned
#define KCHUNK   128                    // K staged per LDS buffer (4 WMMA k-steps)
#define WROW     136                    // padded LDS W-row stride (bf16), 16B-aligned

typedef __attribute__((ext_vector_type(16))) __bf16 v16bf;
typedef __attribute__((ext_vector_type(8)))  __bf16 v8bf;
typedef __attribute__((ext_vector_type(8)))  float  v8f;
typedef int v4i __attribute__((vector_size(16)));   // matches builtin param type

#if __has_builtin(__builtin_amdgcn_global_load_async_to_lds_b128) && \
    __has_builtin(__builtin_amdgcn_s_wait_asynccnt)
#define USE_ASYNC_LDS 1
#else
#define USE_ASYNC_LDS 0
#endif

// 16-byte global -> LDS copy (async DMA path when available).
static __device__ __forceinline__ void async_copy16(const __bf16* g, __bf16* l) {
#if USE_ASYNC_LDS
    __builtin_amdgcn_global_load_async_to_lds_b128(
        (__attribute__((address_space(1))) v4i*)(void*)g,
        (__attribute__((address_space(3))) v4i*)(void*)l, 0, 0);
#else
    *(v8bf*)l = *(const v8bf*)g;
#endif
}

#if USE_ASYNC_LDS
#define WAIT_ASYNC(n) __builtin_amdgcn_s_wait_asynccnt(n)
#else
#define WAIT_ASYNC(n)
#endif

static __device__ __forceinline__ v16bf make_a(v8bf lo, v8bf hi) {
    v16bf a;
#pragma unroll
    for (int i = 0; i < 8; ++i) { a[i] = lo[i]; a[i + 8] = hi[i]; }
    return a;
}

// ---------------------------------------------------------------------------
// fp32 -> bf16 elementwise convert
// ---------------------------------------------------------------------------
__global__ void cvt_bf16_kernel(const float* __restrict__ in,
                                __bf16* __restrict__ out, int n) {
    int i = blockIdx.x * blockDim.x + threadIdx.x;
    if (i < n) out[i] = (__bf16)in[i];
}

__global__ void zero_kernel(float* __restrict__ p, int n) {
    int i = blockIdx.x * blockDim.x + threadIdx.x;
    if (i < n) p[i] = 0.0f;
}

// ---------------------------------------------------------------------------
// Fused QKV projection: Y = X @ W^T + b  (M=8192, N=1024, K=1024), z selects
// Q / K / V.  Block = 256 threads (8 waves), tile 128(M) x 64(N).
// W tiles staged 128-K at a time into double-buffered LDS via async DMA;
// next chunk's DMA overlaps the current chunk's 16 WMMAs.
// Q,K stored bf16 head-major [B,H,S,Hd]; V stored transposed [B,H,Hd,S].
// ---------------------------------------------------------------------------
__global__ void qkv_gemm_kernel(const __bf16* __restrict__ Xbf,
                                const __bf16* __restrict__ Wqbf,
                                const __bf16* __restrict__ Wkbf,
                                const __bf16* __restrict__ Wvbf,
                                const float* __restrict__ bq,
                                const float* __restrict__ bk,
                                const float* __restrict__ bv,
                                __bf16* __restrict__ Qout,
                                __bf16* __restrict__ Kout,
                                __bf16* __restrict__ Vtout) {
    __shared__ __bf16 lds_w[2][64 * WROW];  // double-buffered 64 x 128 bf16 (+pad)

    const int wave = threadIdx.x >> 5;
    const int lane = threadIdx.x & 31;
    const int m0 = blockIdx.x * 128 + wave * 16;
    const int n0 = blockIdx.y * 64;
    const int z  = blockIdx.z;

    const __bf16* W   = (z == 0) ? Wqbf : (z == 1) ? Wkbf : Wvbf;
    const float* bias = (z == 0) ? bq   : (z == 1) ? bk   : bv;

    v8f c0 = {}, c1 = {}, c2 = {}, c3 = {};

    const int nl   = lane & 15;
    const int arow = m0 + nl;                 // A rows: lanes 0-15 == 16-31
    const int ac0  = (lane < 16) ? 0 : 8;     // A K-chunk base per ISA layout
    const int bkb  = (lane < 16) ? 0 : 8;     // B K-chunk base per ISA layout

    // Staging map: thread copies 32 bf16 (4 x 16B) of one W row.
    const int srow = threadIdx.x >> 2;        // 0..63
    const int scol = (threadIdx.x & 3) * 32;  // 0,32,64,96

    const __bf16* wsrc = &W[(size_t)(n0 + srow) * HIDDEN + scol];
    __bf16* wdst0 = &lds_w[0][srow * WROW + scol];
    __bf16* wdst1 = &lds_w[1][srow * WROW + scol];

    // Prefetch chunk 0.
#pragma unroll
    for (int j = 0; j < 4; ++j) async_copy16(wsrc + j * 8, wdst0 + j * 8);

    const int NCHUNK = HIDDEN / KCHUNK;       // 8
    for (int chunk = 0; chunk < NCHUNK; ++chunk) {
        const int kc = chunk * KCHUNK;
        if (chunk + 1 < NCHUNK) {
            const __bf16* s = wsrc + (size_t)(kc + KCHUNK);
            __bf16* d = ((chunk + 1) & 1) ? wdst1 : wdst0;
#pragma unroll
            for (int j = 0; j < 4; ++j) async_copy16(s + j * 8, d + j * 8);
            WAIT_ASYNC(4);                    // current chunk landed; next in flight
        } else {
            WAIT_ASYNC(0);
        }
        __syncthreads();

        const __bf16* lw = lds_w[chunk & 1];
#pragma unroll
        for (int ks = 0; ks < 4; ++ks) {
            const int k = kc + ks * 32;
            v8bf alo = *(const v8bf*)&Xbf[(size_t)arow * HIDDEN + k + ac0];
            v8bf ahi = *(const v8bf*)&Xbf[(size_t)arow * HIDDEN + k + ac0 + 16];
            v16bf a = make_a(alo, ahi);
#pragma unroll
            for (int t = 0; t < 4; ++t) {
                const int nrow = t * 16 + nl;
                v8bf blo = *(const v8bf*)&lw[nrow * WROW + ks * 32 + bkb];
                v8bf bhi = *(const v8bf*)&lw[nrow * WROW + ks * 32 + bkb + 16];
                v16bf b = make_a(blo, bhi);
                v8f* cp = (t == 0) ? &c0 : (t == 1) ? &c1 : (t == 2) ? &c2 : &c3;
                *cp = __builtin_amdgcn_wmma_f32_16x16x32_bf16(
                    false, a, false, b, (short)0, *cp, false, false);
            }
        }
        __syncthreads();
    }

    // Epilogue: bias + scatter to head-major layouts.
#pragma unroll
    for (int t = 0; t < 4; ++t) {
        v8f cc = (t == 0) ? c0 : (t == 1) ? c1 : (t == 2) ? c2 : c3;
        const int ng = n0 + t * 16 + nl;
        const float bb = bias[ng];
        const int h = ng >> 6, d = ng & 63;
#pragma unroll
        for (int r = 0; r < 8; ++r) {
            const int m = m0 + r + ((lane < 16) ? 0 : 8);
            const int bidx = m >> 11;
            const int s = m & (SEQ - 1);
            const __bf16 vb = (__bf16)(cc[r] + bb);
            if (z == 0)
                Qout[(((size_t)bidx * HEADS + h) * SEQ + s) * HEAD_DIM + d] = vb;
            else if (z == 1)
                Kout[(((size_t)bidx * HEADS + h) * SEQ + s) * HEAD_DIM + d] = vb;
            else
                Vtout[(((size_t)bidx * HEADS + h) * HEAD_DIM + d) * SEQ + s] = vb;
        }
    }
}

// ---------------------------------------------------------------------------
// Attention: one block (4 waves, 128 thr) per (b,h, 16-row q-tile).
//  P1: waves split S; energy = (Q K^T)*scale (+mask) into LDS [16 x 2048].
//  P2: cooperative softmax; attn streamed to d_out with NT stores.
//  P3: per-wave 16x16 d-tile of attn @ V via WMMA from LDS; column sums
//      atomically folded into osum[B][1024].
// ---------------------------------------------------------------------------
__global__ void attn_kernel(const __bf16* __restrict__ Qbf,
                            const __bf16* __restrict__ Kbf,
                            const __bf16* __restrict__ Vt,
                            const int* __restrict__ mask,
                            float* __restrict__ attn_out,
                            float* __restrict__ osum) {
    extern __shared__ float smem[];
    float* lds_e   = smem;                 // 16 * EROW
    float* scratch = smem + 16 * EROW;     // 128

    const int wave = threadIdx.x >> 5;
    const int lane = threadIdx.x & 31;
    const int nl   = lane & 15;
    const int ac0  = (lane < 16) ? 0 : 8;
    const int bkb  = (lane < 16) ? 0 : 8;
    const int mro  = (lane < 16) ? 0 : 8;

    const int bh   = blockIdx.y;
    const int bidx = bh >> 4;
    const int h    = bh & 15;
    const int q0   = blockIdx.x * 16;

    // ---- Phase 1: energy -------------------------------------------------
    const size_t qbase = ((size_t)bh * SEQ + q0) * HEAD_DIM + (size_t)nl * HEAD_DIM;
    v16bf a0 = make_a(*(const v8bf*)&Qbf[qbase + ac0],
                      *(const v8bf*)&Qbf[qbase + ac0 + 16]);
    v16bf a1 = make_a(*(const v8bf*)&Qbf[qbase + 32 + ac0],
                      *(const v8bf*)&Qbf[qbase + 32 + ac0 + 16]);

    const __bf16* Kbase = Kbf + (size_t)bh * SEQ * HEAD_DIM;
    for (int kt = wave * 32; kt < wave * 32 + 32; ++kt) {
        const int key0 = kt * 16;
        v8f acc = {};
        const __bf16* p0 = Kbase + (size_t)(key0 + nl) * HEAD_DIM + bkb;
        v16bf b0 = make_a(*(const v8bf*)p0, *(const v8bf*)(p0 + 16));
        acc = __builtin_amdgcn_wmma_f32_16x16x32_bf16(
            false, a0, false, b0, (short)0, acc, false, false);
        const __bf16* p1 = p0 + 32;
        v16bf b1 = make_a(*(const v8bf*)p1, *(const v8bf*)(p1 + 16));
        acc = __builtin_amdgcn_wmma_f32_16x16x32_bf16(
            false, a1, false, b1, (short)0, acc, false, false);

        const int key = key0 + nl;
        const int mv  = mask[bidx * SEQ + key];
#pragma unroll
        for (int r = 0; r < 8; ++r) {
            float e = acc[r] * 0.125f;          // 1/sqrt(64)
            if (mv == 0) e = -10000.0f;
            lds_e[(mro + r) * EROW + key] = e;
        }
    }
    __syncthreads();

    // ---- Phase 2: softmax + NT stream of attn ----------------------------
    const int row = threadIdx.x >> 3;   // 0..15
    const int sub = threadIdx.x & 7;    // 8 threads per row
    float mx = -3.0e38f;
    for (int cix = sub; cix < SEQ; cix += 8)
        mx = fmaxf(mx, lds_e[row * EROW + cix]);
    scratch[row * 8 + sub] = mx;
    __syncthreads();
    float mr = scratch[row * 8];
#pragma unroll
    for (int i = 1; i < 8; ++i) mr = fmaxf(mr, scratch[row * 8 + i]);
    __syncthreads();
    float sm = 0.0f;
    for (int cix = sub; cix < SEQ; cix += 8) {
        float v = __expf(lds_e[row * EROW + cix] - mr);
        lds_e[row * EROW + cix] = v;
        sm += v;
    }
    scratch[row * 8 + sub] = sm;
    __syncthreads();
    float sr = 0.0f;
#pragma unroll
    for (int i = 0; i < 8; ++i) sr += scratch[row * 8 + i];
    const float inv = 1.0f / sr;
    float* arow_out = attn_out + ((size_t)bh * SEQ + (q0 + row)) * SEQ;
    for (int cix = sub; cix < SEQ; cix += 8) {
        float v = lds_e[row * EROW + cix] * inv;
        lds_e[row * EROW + cix] = v;
        __builtin_nontemporal_store(v, &arow_out[cix]);   // 1GB pure stream
    }
    __syncthreads();

    // ---- Phase 3: out_tile = attn @ V (per-wave 16x16 d-tile) ------------
    v8f c = {};
    const __bf16* Vbase = Vt + ((size_t)bh * HEAD_DIM + wave * 16 + nl) * SEQ;
    for (int kk = 0; kk < SEQ / 32; ++kk) {
        v8f f0 = *(const v8f*)&lds_e[nl * EROW + kk * 32 + ac0];
        v8f f1 = *(const v8f*)&lds_e[nl * EROW + kk * 32 + ac0 + 16];
        v16bf a;
#pragma unroll
        for (int i = 0; i < 8; ++i) { a[i] = (__bf16)f0[i]; a[i + 8] = (__bf16)f1[i]; }
        v16bf b = make_a(*(const v8bf*)&Vbase[kk * 32 + bkb],
                         *(const v8bf*)&Vbase[kk * 32 + bkb + 16]);
        c = __builtin_amdgcn_wmma_f32_16x16x32_bf16(
            false, a, false, b, (short)0, c, false, false);
    }
    float partial = 0.0f;
#pragma unroll
    for (int r = 0; r < 8; ++r) partial += c[r];
    atomicAdd(&osum[(size_t)bidx * HIDDEN + h * HEAD_DIM + wave * 16 + nl], partial);
}

// ---------------------------------------------------------------------------
// weighted[b,n] = sum_k osum[b,k] * Wo[n,k] + SEQ * bo[n]
// ---------------------------------------------------------------------------
__global__ void outproj_kernel(const float* __restrict__ osum,
                               const float* __restrict__ Wo,
                               const float* __restrict__ bo,
                               float* __restrict__ weighted) {
    const int tid = blockIdx.x * blockDim.x + threadIdx.x;
    if (tid >= BATCH * HIDDEN) return;
    const int b = tid >> 10, n = tid & (HIDDEN - 1);
    const float* os = osum + (size_t)b * HIDDEN;
    const float* w  = Wo + (size_t)n * HIDDEN;
    float acc = 0.0f;
    for (int k = 0; k < HIDDEN; k += 4)
        acc += os[k] * w[k] + os[k + 1] * w[k + 1] +
               os[k + 2] * w[k + 2] + os[k + 3] * w[k + 3];
    weighted[tid] = acc + (float)SEQ * bo[n];
}

// ---------------------------------------------------------------------------
extern "C" void kernel_launch(void* const* d_in, const int* in_sizes, int n_in,
                              void* d_out, int out_size, void* d_ws, size_t ws_size,
                              hipStream_t stream) {
    const float* X    = (const float*)d_in[0];
    const int*   mask = (const int*)d_in[1];
    const float* Wq   = (const float*)d_in[2];
    const float* bq   = (const float*)d_in[3];
    const float* Wk   = (const float*)d_in[4];
    const float* bk   = (const float*)d_in[5];
    const float* Wv   = (const float*)d_in[6];
    const float* bv   = (const float*)d_in[7];
    const float* Wo   = (const float*)d_in[8];
    const float* bo   = (const float*)d_in[9];
    float* out = (float*)d_out;

    char* ws = (char*)d_ws;
    size_t off = 0;
    __bf16* Xbf  = (__bf16*)(ws + off); off += (size_t)MROWS * HIDDEN * 2;
    __bf16* Wqbf = (__bf16*)(ws + off); off += (size_t)HIDDEN * HIDDEN * 2;
    __bf16* Wkbf = (__bf16*)(ws + off); off += (size_t)HIDDEN * HIDDEN * 2;
    __bf16* Wvbf = (__bf16*)(ws + off); off += (size_t)HIDDEN * HIDDEN * 2;
    __bf16* Qbf  = (__bf16*)(ws + off); off += (size_t)MROWS * HIDDEN * 2;
    __bf16* Kbf  = (__bf16*)(ws + off); off += (size_t)MROWS * HIDDEN * 2;
    __bf16* Vt   = (__bf16*)(ws + off); off += (size_t)MROWS * HIDDEN * 2;
    float*  osum = (float*)(ws + off);  off += (size_t)BATCH * HIDDEN * 4;

    const int nX = MROWS * HIDDEN;
    const int nW = HIDDEN * HIDDEN;
    cvt_bf16_kernel<<<(nX + 255) / 256, 256, 0, stream>>>(X,  Xbf,  nX);
    cvt_bf16_kernel<<<(nW + 255) / 256, 256, 0, stream>>>(Wq, Wqbf, nW);
    cvt_bf16_kernel<<<(nW + 255) / 256, 256, 0, stream>>>(Wk, Wkbf, nW);
    cvt_bf16_kernel<<<(nW + 255) / 256, 256, 0, stream>>>(Wv, Wvbf, nW);
    zero_kernel<<<(BATCH * HIDDEN + 255) / 256, 256, 0, stream>>>(osum, BATCH * HIDDEN);

    dim3 gA(MROWS / 128, HIDDEN / 64, 3);
    qkv_gemm_kernel<<<gA, 256, 0, stream>>>(Xbf, Wqbf, Wkbf, Wvbf,
                                            bq, bk, bv, Qbf, Kbf, Vt);

    dim3 gB(SEQ / 16, BATCH * HEADS);
    const size_t shmem = (size_t)(16 * EROW + 128) * sizeof(float); // ~132 KB of 320 KB LDS
    attn_kernel<<<gB, 128, shmem, stream>>>(Qbf, Kbf, Vt, mask, out + 4096, osum);

    outproj_kernel<<<(BATCH * HIDDEN + 255) / 256, 256, 0, stream>>>(osum, Wo, bo, out);
}
